// EmbeddingLoss_65094524338844
// MI455X (gfx1250) — matrix operand order repaired and checked by reading, not compiled, not run on hardware
//
#include <hip/hip_runtime.h>
#include <hip/hip_bf16.h>
#include <cstdint>

// ---------------------------------------------------------------------------
// CDNA5 / gfx1250 implementation of the UM embedding loss pipeline.
//  Phase 1: all-pairs distance matrix via v_wmma_f32_16x16x32_bf16 with
//           TDM tensor_load_to_lds staging of the operand panels
//  Phase 2: serial Prim MST, one persistent workgroup, bf16 rows from HBM
//  Phase 3: bitonic sort of edge distances (single block, global memory)
//  Phase 4: serial Kruskal merge with 64-label component histograms
//  Phase 5: output writeback
// Workspace requirement: >= 0x800000 + N*N*2 bytes  (~520 MiB)
// ---------------------------------------------------------------------------

typedef __attribute__((ext_vector_type(16))) __bf16 v16bf;
typedef __attribute__((ext_vector_type(8)))  float  v8f;
typedef __attribute__((ext_vector_type(4)))  unsigned int u32x4;
typedef __attribute__((ext_vector_type(4)))  int   i32x4;
typedef __attribute__((ext_vector_type(8)))  int   i32x8;

#define NPTS   16384
#define NEDGE  16383
#define KPAD   32          // K padded to one WMMA's worth of bf16
#define NLAB   64
#define ALPHA  0.5f
#define CSCALE 0.01f
#define INF_F  3.0e38f

#define CT_PER_WAVE   4            // column tiles per wave
#define COLS_PER_BLK  512          // 8 waves * 4 tiles * 16 cols

// ---------------------------------------------------------------------------
// TDM 2-D tile load: rows x 16-dword lines from ptsb (row stride 16 dwords)
// into LDS.  D# built per CDNA5 ISA ch.8: group0 = {count, lds, gaddr, type},
// group1 = {data_size, tensor dims, tile dims, strides}; groups 2/3 zero
// (2-D tensor, VADDR2/3 semantics disabled).
// ---------------------------------------------------------------------------
__device__ __forceinline__ void tdm_load_2d(unsigned lds_addr, const void* gptr,
                                            unsigned rows) {
  unsigned long long ga = (unsigned long long)(uintptr_t)gptr;
  u32x4 g0;
  g0[0] = 1u;                                      // count=1, user descriptor
  g0[1] = lds_addr;                                // LDS byte address
  g0[2] = (unsigned)(ga & 0xffffffffu);            // global_addr[31:0]
  g0[3] = (unsigned)((ga >> 32) & 0x01ffffffu)     // global_addr[56:32]
        | (2u << 30);                              // type = 2 ("image")
  i32x8 g1;
  g1[0] = (int)(2u << 16);                         // data_size = 4B, mask=0
  g1[1] = (int)(16u << 16);                        // tensor_dim0 = 16 dwords
  g1[2] = (int)(((unsigned)NPTS & 0xffffu) << 16); // tensor_dim1 lo16
  g1[3] = (int)((((unsigned)NPTS >> 16) & 0xffffu)
        | (16u << 16));                            // dim1 hi16 | tile_dim0=16
  g1[4] = (int)(rows & 0xffffu);                   // tile_dim1 = rows (2-D)
  g1[5] = 16;                                      // tensor_dim0_stride lo32
  g1[6] = 0;                                       // stride hi | dim1_stride lo
  g1[7] = 0;
  i32x4 z4 = {0, 0, 0, 0};
#if defined(__clang_major__) && (__clang_major__ >= 23)
  i32x8 z8 = {0, 0, 0, 0, 0, 0, 0, 0};
  __builtin_amdgcn_tensor_load_to_lds(g0, g1, z4, z4, z8, 0);
#else
  __builtin_amdgcn_tensor_load_to_lds(g0, g1, z4, z4, 0);
#endif
}

// ---------------------------------------------------------------------------
// Phase 0: build points [N,8] f32 (padded), bf16 copy [N,32] (K padded to 32
// for one-shot WMMA), and squared norms.
// ---------------------------------------------------------------------------
__global__ void build_points_kernel(const float* __restrict__ emb,
                                    float* __restrict__ pts,
                                    __bf16* __restrict__ ptsb,
                                    float* __restrict__ norms) {
  int i = blockIdx.x * blockDim.x + threadIdx.x;
  if (i >= NPTS) return;
  int w = i & 31, h = (i >> 5) & 31, d = i >> 10;
  float p[8];
  p[0] = emb[0 * NPTS + i];
  p[1] = emb[1 * NPTS + i];
  p[2] = emb[2 * NPTS + i];
  p[3] = CSCALE * (float)d;
  p[4] = CSCALE * (float)h;
  p[5] = CSCALE * (float)w;
  p[6] = 0.f; p[7] = 0.f;
  float n = 0.f;
#pragma unroll
  for (int k = 0; k < 8; ++k) { pts[i * 8 + k] = p[k]; n += p[k] * p[k]; }
  norms[i] = n;
#pragma unroll
  for (int k = 0; k < KPAD; ++k)
    ptsb[(size_t)i * KPAD + k] = (k < 6) ? (__bf16)p[k] : (__bf16)0.f;
}

// ---------------------------------------------------------------------------
// Phase 1: dist(i,j) strip per block: 16 rows x 512 cols.
//  - wave 0 issues two TDM loads (A tile 1 KiB, B panel 32 KiB) into LDS,
//    s_wait_tensorcnt 0, then workgroup barrier
//  - 8 waves x 4 WMMAs each: G = A*B^T, dist = sqrt(ni + nj - 2G) -> bf16
// A (16x32 bf16) per ISA layout: lanes 0-15 hold K 0-7/16-23, lanes 16-31
// hold K 8-15/24-31, M = lane&15.  B (32x16): same K order as A,
// column = lane&15.
// ---------------------------------------------------------------------------
__global__ __launch_bounds__(256)
void dist_tiles_kernel(const __bf16* __restrict__ ptsb,
                       const float*  __restrict__ norms,
                       __bf16* __restrict__ dmat) {
  __shared__ __bf16 sA[16 * KPAD];             //  1 KiB
  __shared__ __bf16 sB[COLS_PER_BLK * KPAD];   // 32 KiB
  const int bi = blockIdx.x & 1023;            // row tile   (1024)
  const int cg = blockIdx.x >> 10;             // col group  (32)
  const int i0 = bi << 4;
  const int c0 = cg * COLS_PER_BLK;
  const int t = threadIdx.x;
  const int wid = t >> 5, lane = t & 31;
  const int half = lane >> 4, mr = lane & 15;

  if (wid == 0) {                              // one wave drives the TDM
    tdm_load_2d((unsigned)(uintptr_t)(void*)sA,
                (const void*)(ptsb + (size_t)i0 * KPAD), 16u);
    tdm_load_2d((unsigned)(uintptr_t)(void*)sB,
                (const void*)(ptsb + (size_t)c0 * KPAD), (unsigned)COLS_PER_BLK);
    __builtin_amdgcn_s_wait_tensorcnt(0);
  }
  __syncthreads();

  // A operand (shared by all 4 column tiles of this wave)
  v16bf a;
  const __bf16* arow = sA + mr * KPAD;
#pragma unroll
  for (int v = 0; v < 4; ++v) {
    a[2 * v]     = arow[half * 8 + 2 * v];
    a[2 * v + 1] = arow[half * 8 + 2 * v + 1];
    a[2 * v + 8] = arow[16 + half * 8 + 2 * v];
    a[2 * v + 9] = arow[16 + half * 8 + 2 * v + 1];
  }
  float ni[8];
#pragma unroll
  for (int r = 0; r < 8; ++r) ni[r] = norms[i0 + r + half * 8];

#pragma unroll
  for (int ct = 0; ct < CT_PER_WAVE; ++ct) {
    const int jt = (wid * CT_PER_WAVE + ct) << 4;    // col offset in block
    const int j0 = c0 + jt;
    const __bf16* brow = sB + (size_t)(jt + mr) * KPAD;
    v16bf b;
#pragma unroll
    for (int v = 0; v < 4; ++v) {
      b[2 * v]     = brow[half * 8 + 2 * v];
      b[2 * v + 1] = brow[half * 8 + 2 * v + 1];
      b[2 * v + 8] = brow[16 + half * 8 + 2 * v];
      b[2 * v + 9] = brow[16 + half * 8 + 2 * v + 1];
    }
    v8f c = {};
    // (neg_a, A, neg_b, B, c_mod, C, reuse_a, reuse_b)
    c = __builtin_amdgcn_wmma_f32_16x16x32_bf16(false, a, false, b,
                                                (short)0, c, false, false);
    float nj = norms[j0 + mr];
#pragma unroll
    for (int r = 0; r < 8; ++r) {
      float d2 = ni[r] + nj - 2.0f * c[r];
      float dd = sqrtf(fmaxf(d2, 0.f));
      dmat[(size_t)(i0 + r + half * 8) * NPTS + (j0 + mr)] = (__bf16)dd;
    }
  }
}

// ---------------------------------------------------------------------------
// Phase 2: serial Prim MST. 1024 threads; thread t owns points i = k*1024+t
// (k=0..15) entirely in registers. Per step: stream bf16 row j (coalesced),
// wave32 shuffle argmin + LDS cross-wave reduce.
// ---------------------------------------------------------------------------
__global__ __launch_bounds__(1024)
void prim_kernel(const __bf16* __restrict__ dmat,
                 int* __restrict__ usA, int* __restrict__ vsA,
                 float* __restrict__ dsA) {
  const int t = threadIdx.x;
  float md[16]; int mf[16]; unsigned vis = 0u;
#pragma unroll
  for (int k = 0; k < 16; ++k) {
    int i = (k << 10) + t;
    md[k] = (float)dmat[i];         // distances from point 0 (row 0)
    mf[k] = 0;
  }
  if (t == 0) { md[0] = INF_F; vis |= 1u; }   // point 0 visited

  __shared__ float rd[32];
  __shared__ int   ri[32];
  __shared__ int   bjS;
  __shared__ float bdS;
  const int lane = t & 31, wid = t >> 5;

  for (int step = 0; step < NEDGE; ++step) {
    float bd = INF_F; int bi_ = 0x7fffffff;
#pragma unroll
    for (int k = 0; k < 16; ++k)
      if (md[k] < bd) { bd = md[k]; bi_ = (k << 10) + t; }
    // wave32 argmin
#pragma unroll
    for (int off = 16; off > 0; off >>= 1) {
      float od = __shfl_down(bd, off, 32);
      int   oi = __shfl_down(bi_, off, 32);
      if (od < bd || (od == bd && oi < bi_)) { bd = od; bi_ = oi; }
    }
    if (lane == 0) { rd[wid] = bd; ri[wid] = bi_; }
    __syncthreads();
    if (wid == 0) {
      bd = rd[lane]; bi_ = ri[lane];
#pragma unroll
      for (int off = 16; off > 0; off >>= 1) {
        float od = __shfl_down(bd, off, 32);
        int   oi = __shfl_down(bi_, off, 32);
        if (od < bd || (od == bd && oi < bi_)) { bd = od; bi_ = oi; }
      }
      if (lane == 0) { bjS = bi_; bdS = bd; }
    }
    __syncthreads();
    const int j = bjS; const float dj = bdS;
    if ((j & 1023) == t) {                       // owner records the edge
      int k = j >> 10;
      usA[step] = mf[k]; vsA[step] = j; dsA[step] = dj;
      vis |= (1u << k); md[k] = INF_F;
    }
    const __bf16* row = dmat + (size_t)j * NPTS; // stream row j (32 KB)
#pragma unroll
    for (int k = 0; k < 16; ++k) {
      int i = (k << 10) + t;
      float f = (float)row[i];
      if (!((vis >> k) & 1u) && f < md[k]) { md[k] = f; mf[k] = j; }
    }
    __syncthreads();
  }
}

// ---------------------------------------------------------------------------
// Phase 3: bitonic sort of 16384 keys (edge NEDGE padded with +inf), single
// block, arrays in global memory (L2 resident: 128 KB working set).
// ---------------------------------------------------------------------------
__global__ __launch_bounds__(1024)
void sort_kernel(const float* __restrict__ dsA,
                 float* __restrict__ skey, int* __restrict__ sidx) {
  const int t = threadIdx.x;
  for (int i = t; i < NPTS; i += 1024) {
    skey[i] = (i < NEDGE) ? dsA[i] : INF_F;
    sidx[i] = i;
  }
  __syncthreads();
  for (int k = 2; k <= NPTS; k <<= 1) {
    for (int j = k >> 1; j > 0; j >>= 1) {
      for (int i = t; i < NPTS; i += 1024) {
        int l = i ^ j;
        if (l > i) {
          bool up = ((i & k) == 0);
          float a = skey[i], b = skey[l];
          if ((a > b) == up) {
            skey[i] = b; skey[l] = a;
            int tmp = sidx[i]; sidx[i] = sidx[l]; sidx[l] = tmp;
          }
        }
      }
      __syncthreads();
    }
  }
}

// ---------------------------------------------------------------------------
// Phase 4: serial Kruskal merge over sorted MST edges. 64-lane-parallel
// histogram dot + merge, 256-thread-parallel component relabel. Also computes
// pos/neg sums and the scalar loss.
// ---------------------------------------------------------------------------
__global__ __launch_bounds__(256)
void kruskal_kernel(const int* __restrict__ target, const float* __restrict__ mask,
                    const int* __restrict__ usA, const int* __restrict__ vsA,
                    const float* __restrict__ skey, const int* __restrict__ sidx,
                    int* __restrict__ comp, float* __restrict__ cnts,
                    float* __restrict__ csize,
                    float* __restrict__ posA, float* __restrict__ negA,
                    float* __restrict__ sums, float* __restrict__ out) {
  const int t = threadIdx.x;
  for (int i = t; i < NPTS; i += 256) {
    comp[i] = i;
    float m = mask[i];
    csize[i] = m;
    int lab = target[i] & (NLAB - 1);
    for (int l = 0; l < NLAB; ++l)
      cnts[(size_t)i * NLAB + l] = (l == lab) ? m : 0.f;
  }
  __shared__ float part[256];
  __shared__ int cuS, cvS;
  __syncthreads();

  for (int e = 0; e < NEDGE; ++e) {
    if (t == 0) {
      int x = sidx[e];
      cuS = comp[usA[x]];
      cvS = comp[vsA[x]];
    }
    __syncthreads();
    const int cu = cuS, cv = cvS;
    float p = 0.f;
    if (t < NLAB) p = cnts[(size_t)cu * NLAB + t] * cnts[(size_t)cv * NLAB + t];
    part[t] = p;
    __syncthreads();
    for (int off = 128; off > 0; off >>= 1) {
      if (t < off) part[t] += part[t + off];
      __syncthreads();
    }
    if (t == 0) {
      float pos = part[0];
      float tot = csize[cu] * csize[cv];
      posA[e] = pos; negA[e] = tot - pos;
      csize[cu] += csize[cv];
    }
    if (t < NLAB) cnts[(size_t)cu * NLAB + t] += cnts[(size_t)cv * NLAB + t];
    for (int i = t; i < NPTS; i += 256)
      if (comp[i] == cv) comp[i] = cu;
    __syncthreads();
  }

  // sums and loss
  float sp = 0.f, sn = 0.f;
  for (int e = t; e < NEDGE; e += 256) { sp += posA[e]; sn += negA[e]; }
  part[t] = sp; __syncthreads();
  for (int off = 128; off > 0; off >>= 1) { if (t < off) part[t] += part[t + off]; __syncthreads(); }
  float Sp = fmaxf(part[0], 1.0f); __syncthreads();
  part[t] = sn; __syncthreads();
  for (int off = 128; off > 0; off >>= 1) { if (t < off) part[t] += part[t + off]; __syncthreads(); }
  float Sn = fmaxf(part[0], 1.0f); __syncthreads();

  float ll = 0.f;
  for (int e = t; e < NEDGE; e += 256) {
    float d = skey[e];
    float hinge = fmaxf(ALPHA - d, 0.f);
    ll += posA[e] / Sp * d * d + negA[e] / Sn * hinge * hinge;
  }
  part[t] = ll; __syncthreads();
  for (int off = 128; off > 0; off >>= 1) { if (t < off) part[t] += part[t + off]; __syncthreads(); }
  if (t == 0) { out[0] = part[0]; sums[0] = Sp; sums[1] = Sn; }
}

// ---------------------------------------------------------------------------
// Phase 5: grid-parallel output writeback.
// d_out layout: [loss(1), emst(E*3), edges_u(E*6), edges_v(E*6), ds(E),
//                ratio_pos(E), ratio_neg(E)]
// ---------------------------------------------------------------------------
__global__ void finalize_kernel(const float* __restrict__ pts,
                                const int* __restrict__ usA, const int* __restrict__ vsA,
                                const float* __restrict__ skey, const int* __restrict__ sidx,
                                const float* __restrict__ posA, const float* __restrict__ negA,
                                const float* __restrict__ sums, float* __restrict__ out) {
  int s = blockIdx.x * blockDim.x + threadIdx.x;
  if (s >= NEDGE) return;
  int x = sidx[s];
  int u = usA[x], v = vsA[x];
  float d = skey[s];
  const int b1 = 1 + NEDGE * 3;
  const int b2 = b1 + NEDGE * 6;
  const int b3 = b2 + NEDGE * 6;
  const int b4 = b3 + NEDGE;
  const int b5 = b4 + NEDGE;
  out[1 + s * 3 + 0] = (float)u;
  out[1 + s * 3 + 1] = (float)v;
  out[1 + s * 3 + 2] = d;
#pragma unroll
  for (int c = 0; c < 6; ++c) {
    out[b1 + s * 6 + c] = pts[u * 8 + c];
    out[b2 + s * 6 + c] = pts[v * 8 + c];
  }
  out[b3 + s] = d;
  out[b4 + s] = posA[s] / sums[0];
  out[b5 + s] = negA[s] / sums[1];
}

// ---------------------------------------------------------------------------
extern "C" void kernel_launch(void* const* d_in, const int* in_sizes, int n_in,
                              void* d_out, int out_size, void* d_ws, size_t ws_size,
                              hipStream_t stream) {
  const float* emb    = (const float*)d_in[0];   // [3,16,32,32] f32
  const int*   target = (const int*)d_in[1];     // [16384] int
  const float* mask   = (const float*)d_in[2];   // [16384] f32
  float* out = (float*)d_out;

  char* w = (char*)d_ws;
  float*  pts   = (float*) (w + 0x000000);  // N*8*4   = 512 KiB
  __bf16* ptsb  = (__bf16*)(w + 0x080000);  // N*32*2  = 1 MiB
  float*  norms = (float*) (w + 0x180000);  // 64 KiB
  int*    usA   = (int*)   (w + 0x190000);
  int*    vsA   = (int*)   (w + 0x1A0000);
  float*  dsA   = (float*) (w + 0x1B0000);
  float*  skey  = (float*) (w + 0x1C0000);
  int*    sidx  = (int*)   (w + 0x1D0000);
  int*    comp  = (int*)   (w + 0x1E0000);
  float*  csize = (float*) (w + 0x1F0000);
  float*  posA  = (float*) (w + 0x200000);
  float*  negA  = (float*) (w + 0x210000);
  float*  sums  = (float*) (w + 0x220000);
  float*  cnts  = (float*) (w + 0x400000);  // N*64*4 = 4 MiB
  __bf16* dmat  = (__bf16*)(w + 0x800000);  // N*N*2  = 512 MiB

  build_points_kernel<<<dim3(NPTS / 256), dim3(256), 0, stream>>>(emb, pts, ptsb, norms);
  // 1024 row tiles x 32 column groups, each block = 16 rows x 512 cols
  dist_tiles_kernel<<<dim3(1024 * 32), dim3(256), 0, stream>>>(ptsb, norms, dmat);
  prim_kernel<<<dim3(1), dim3(1024), 0, stream>>>(dmat, usA, vsA, dsA);
  sort_kernel<<<dim3(1), dim3(1024), 0, stream>>>(dsA, skey, sidx);
  kruskal_kernel<<<dim3(1), dim3(256), 0, stream>>>(target, mask, usA, vsA, skey, sidx,
                                                    comp, cnts, csize, posA, negA, sums, out);
  finalize_kernel<<<dim3(64), dim3(256), 0, stream>>>(pts, usA, vsA, skey, sidx,
                                                      posA, negA, sums, out);
}
